// Attention_37074157699663
// MI455X (gfx1250) — matrix operand order repaired and checked by reading, not compiled, not run on hardware
//
#include <hip/hip_runtime.h>
#include <math.h>

#define B_DIM 64
#define T_DIM 2048
#define D_DIM 512
#define M_DIM 512
#define INV_SQRT_2PI 0.3989422804014327f

typedef __attribute__((ext_vector_type(16))) __bf16 v16bf;
typedef __attribute__((ext_vector_type(8)))  float  v8f;

// -------------------------------------------------------------------------
// Kernel 1: comp_dec[b,m] = relu(dec[b,:] . phi_w[m,:] + phi_b[m])
// Tiny (64x512 output); plain fp32 dot products.
// -------------------------------------------------------------------------
__global__ __launch_bounds__(256) void comp_dec_kernel(
    const float* __restrict__ dec, const float* __restrict__ phi_w,
    const float* __restrict__ phi_b, float* __restrict__ comp_dec) {
  int idx = blockIdx.x * blockDim.x + threadIdx.x;
  if (idx >= B_DIM * M_DIM) return;
  int b = idx / M_DIM, m = idx % M_DIM;
  const float* x = dec + (size_t)b * D_DIM;
  const float* w = phi_w + (size_t)m * D_DIM;
  float s = phi_b[m];
  for (int d = 0; d < D_DIM; ++d) s = fmaf(x[d], w[d], s);
  comp_dec[idx] = s > 0.f ? s : 0.f;
}

// -------------------------------------------------------------------------
// Kernel 2 (the big one): fused  energy[b,t] = sum_m comp_dec[b,m] *
//     relu(lis[b,t,:] . psi_w[m,:] + psi_b[m])
// One wave (32 lanes) per (b, 16-row t-tile). WMMA bf16 16x16x32, f32 acc.
//   A tile: 16 t-rows x K=32 of lis        (M dim = t)
//   B tile: K=32 x 16 m-cols of psi_w^T    (N dim = m; B col = psi_w row)
// Epilogue per m-tile applied in-register on the C/D fragment; reduction
// over m = cross-lane butterfly within each 16-lane half of the wave32.
// -------------------------------------------------------------------------
__global__ __launch_bounds__(32) void energy_kernel(
    const float* __restrict__ lis,       // (B,T,D)
    const float* __restrict__ psi_w,     // (M,D)
    const float* __restrict__ psi_b,     // (M)
    const float* __restrict__ comp_dec,  // (B,M)
    float* __restrict__ energy)          // (B,T)
{
  const int lane = threadIdx.x;   // 0..31 (wave32)
  const int half = lane >> 4;     // 0 | 1
  const int l16  = lane & 15;
  const int tile = blockIdx.x;    // b * (T/16) + t_tile
  const int b     = tile / (T_DIM / 16);
  const int tbase = (tile % (T_DIM / 16)) * 16;

  // ---- Preload & convert all A fragments for this 16xD slab of lis ----
  // A 16x32 bf16 layout: lane half 0 -> K {0..7, 16..23}; half 1 -> K {8..15, 24..31}
  v16bf afrag[D_DIM / 32];
  const float* arow = lis + ((size_t)b * T_DIM + (tbase + l16)) * D_DIM;
#pragma unroll
  for (int kc = 0; kc < D_DIM / 32; ++kc) {
    const int k0 = kc * 32 + half * 8;        // first  group of 8 K's
    const int k1 = kc * 32 + 16 + half * 8;   // second group of 8 K's
    v16bf a;
#pragma unroll
    for (int i = 0; i < 8; ++i) {
      a[i]     = (__bf16)arow[k0 + i];
      a[8 + i] = (__bf16)arow[k1 + i];
    }
    afrag[kc] = a;
  }

  float acc[8];
#pragma unroll
  for (int r = 0; r < 8; ++r) acc[r] = 0.f;

  for (int mt = 0; mt < M_DIM / 16; ++mt) {
    const int mbase = mt * 16;
    // B column N = l16  <=>  psi_w row (mbase + l16)
    const float* brow = psi_w + (size_t)(mbase + l16) * D_DIM;
    if (mt + 1 < M_DIM / 16)
      __builtin_prefetch(psi_w + (size_t)(mbase + 16 + l16) * D_DIM, 0, 1);

    v8f c = {};
#pragma unroll
    for (int kc = 0; kc < D_DIM / 32; ++kc) {
      // B 32x16 bf16 layout: half 0 -> K {0..7, 16..23}; half 1 -> K {8..15, 24..31}
      const int kb = kc * 32 + half * 8;
      v16bf bf;
#pragma unroll
      for (int i = 0; i < 16; ++i) bf[i] = (__bf16)brow[kb + (i & 7) + ((i >> 3) << 4)];
      c = __builtin_amdgcn_wmma_f32_16x16x32_bf16(
              false, afrag[kc], false, bf, (short)0, c, false, false);
    }

    // Epilogue: this lane's m column
    const float bias = psi_b[mbase + l16];
    const float w    = comp_dec[(size_t)b * M_DIM + mbase + l16];
#pragma unroll
    for (int r = 0; r < 8; ++r) {
      float v = c[r] + bias;          // C/D: lane<16 -> t=tbase+r ; lane>=16 -> t=tbase+r+8
      v = v > 0.f ? v : 0.f;          // relu
      acc[r] = fmaf(v, w, acc[r]);    // * comp_dec, accumulate over m
    }
  }

  // Reduce over m (N dim) = 16 lanes within each half of the wave
#pragma unroll
  for (int r = 0; r < 8; ++r) {
    float v = acc[r];
#pragma unroll
    for (int off = 1; off < 16; off <<= 1) v += __shfl_xor(v, off, 32);
    acc[r] = v;
  }
  if (l16 == 0) {
#pragma unroll
    for (int r = 0; r < 8; ++r)
      energy[(size_t)b * T_DIM + tbase + r + half * 8] = acc[r];
  }
}

// -------------------------------------------------------------------------
// Kernel 3: per-column stats over B: lag-1 correlation r[t], col-zero flag.
// -------------------------------------------------------------------------
__global__ __launch_bounds__(256) void stats_kernel(
    const float* __restrict__ e, float* __restrict__ r_out, int* __restrict__ flag) {
  int t = blockIdx.x * blockDim.x + threadIdx.x;
  if (t >= T_DIM) return;
  const bool has_next = (t < T_DIM - 1);
  bool allz = true;
  float s0 = 0.f, s1 = 0.f;
  for (int b = 0; b < B_DIM; ++b) {
    float v0 = e[(size_t)b * T_DIM + t];
    allz = allz && (v0 == 0.f);
    s0 += v0;
    if (has_next) s1 += e[(size_t)b * T_DIM + t + 1];
  }
  if (allz) atomicOr(flag, 1);
  if (!has_next) return;
  const float m0 = s0 * (1.f / B_DIM), m1 = s1 * (1.f / B_DIM);
  float ss0 = 0.f, ss1 = 0.f, num = 0.f;
  for (int b = 0; b < B_DIM; ++b) {
    float c0 = e[(size_t)b * T_DIM + t] - m0;
    float c1 = e[(size_t)b * T_DIM + t + 1] - m1;
    ss0 = fmaf(c0, c0, ss0);
    ss1 = fmaf(c1, c1, ss1);
    num = fmaf(c0, c1, num);
  }
  r_out[t] = num / sqrtf(ss0 * ss1);
}

// -------------------------------------------------------------------------
// Kernel 4: copula density, both softmaxes over T, selection, counts.
// One block (256 threads) per batch row b.
// -------------------------------------------------------------------------
__device__ __forceinline__ float block_reduce_max(float v, float* s) {
  int tid = threadIdx.x;
  s[tid] = v; __syncthreads();
  for (int off = 128; off > 0; off >>= 1) {
    if (tid < off) s[tid] = fmaxf(s[tid], s[tid + off]);
    __syncthreads();
  }
  float r = s[0]; __syncthreads();
  return r;
}
__device__ __forceinline__ float block_reduce_sum(float v, float* s) {
  int tid = threadIdx.x;
  s[tid] = v; __syncthreads();
  for (int off = 128; off > 0; off >>= 1) {
    if (tid < off) s[tid] += s[tid + off];
    __syncthreads();
  }
  float r = s[0]; __syncthreads();
  return r;
}

__global__ __launch_bounds__(256) void attn_kernel(
    const float* __restrict__ e, const float* __restrict__ r,
    const int* __restrict__ flag, float* __restrict__ f1buf,
    float* __restrict__ out /* d_out base */) {
  const int b = blockIdx.x, tid = threadIdx.x;
  __shared__ float sred[256];
  const float* eb = e + (size_t)b * T_DIM;
  float* f1b = f1buf + (size_t)b * T_DIM;

  float maxE = -3.4e38f, maxF = -3.4e38f;
  for (int t = tid; t < T_DIM; t += 256) {
    float ev = eb[t];
    float marg = INV_SQRT_2PI * expf(-0.5f * ev * ev);
    float cp = 1.0f;
    if (t >= 1 && t <= T_DIM - 2) {
      float rr = r[t - 1];                   // rr index = t-1
      float det = 1.0f - rr * rr;
      bool small_det = det < 0.01f;
      float sd = small_det ? 1.0f : det;
      float x1 = eb[t - 1], x2 = ev;
      float expo = (2.0f * rr * x1 * x2 - rr * rr * (x1 * x1 + x2 * x2)) / (2.0f * sd);
      cp = small_det ? 10.0f : (expf(expo) / sqrtf(sd));
    }
    float f1 = marg * cp;
    f1b[t] = f1;
    maxE = fmaxf(maxE, ev);
    maxF = fmaxf(maxF, f1);
  }
  maxE = block_reduce_max(maxE, sred);
  maxF = block_reduce_max(maxF, sred);

  float sumE = 0.f, sumF = 0.f;
  for (int t = tid; t < T_DIM; t += 256) {
    sumE += expf(eb[t] - maxE);
    sumF += expf(f1b[t] - maxF);
  }
  sumE = block_reduce_sum(sumE, sred);
  sumF = block_reduce_sum(sumF, sred);

  const bool use_orig = (*flag != 0);   // B=64 > 3, so only col_zero matters
  const float invE = 1.f / sumE, invF = 1.f / sumF;
  for (int t = tid; t < T_DIM; t += 256) {
    float v = use_orig ? expf(eb[t] - maxE) * invE : expf(f1b[t] - maxF) * invF;
    out[(size_t)b * T_DIM + t] = v;
  }
  if (b == 0 && tid == 0) {
    int* counts = (int*)(out + (size_t)B_DIM * T_DIM + (size_t)B_DIM * D_DIM);
    counts[0] = use_orig ? 1 : 0;   // count_original
    counts[1] = use_orig ? 0 : 1;   // count_copula
  }
}

// -------------------------------------------------------------------------
// Kernel 5: context[b,d] = sum_t lis[b,t,d] * att[b,t]
// -------------------------------------------------------------------------
__global__ __launch_bounds__(256) void context_kernel(
    const float* __restrict__ lis, const float* __restrict__ att,
    float* __restrict__ ctx) {
  const int b = blockIdx.y;
  const int d = blockIdx.x * blockDim.x + threadIdx.x;
  if (d >= D_DIM) return;
  const float* L = lis + (size_t)b * T_DIM * D_DIM;
  const float* A = att + (size_t)b * T_DIM;
  float s = 0.f;
  for (int t = 0; t < T_DIM; ++t) s = fmaf(L[(size_t)t * D_DIM + d], A[t], s);
  ctx[(size_t)b * D_DIM + d] = s;
}

// -------------------------------------------------------------------------
extern "C" void kernel_launch(void* const* d_in, const int* in_sizes, int n_in,
                              void* d_out, int out_size, void* d_ws, size_t ws_size,
                              hipStream_t stream) {
  const float* dec   = (const float*)d_in[0];  // (B,1,D)
  const float* lis   = (const float*)d_in[1];  // (B,T,D)
  const float* phi_w = (const float*)d_in[2];  // (M,D)
  const float* phi_b = (const float*)d_in[3];  // (M)
  const float* psi_w = (const float*)d_in[4];  // (M,D)
  const float* psi_b = (const float*)d_in[5];  // (M)
  float* out = (float*)d_out;                  // [att B*T | ctx B*D | 2 int32]

  float* ws       = (float*)d_ws;
  float* comp_dec = ws;                                         // B*M
  float* energy   = comp_dec + B_DIM * M_DIM;                   // B*T
  float* rbuf     = energy + (size_t)B_DIM * T_DIM;             // T
  float* f1buf    = rbuf + T_DIM;                               // B*T
  int*   flag     = (int*)(f1buf + (size_t)B_DIM * T_DIM);      // 1

  hipMemsetAsync(flag, 0, sizeof(int), stream);

  comp_dec_kernel<<<(B_DIM * M_DIM + 255) / 256, 256, 0, stream>>>(
      dec, phi_w, phi_b, comp_dec);

  energy_kernel<<<B_DIM * (T_DIM / 16), 32, 0, stream>>>(
      lis, psi_w, psi_b, comp_dec, energy);

  stats_kernel<<<(T_DIM + 255) / 256, 256, 0, stream>>>(energy, rbuf, flag);

  attn_kernel<<<B_DIM, 256, 0, stream>>>(energy, rbuf, flag, f1buf, out);

  context_kernel<<<dim3(D_DIM / 256, B_DIM), 256, 0, stream>>>(
      lis, out, out + (size_t)B_DIM * T_DIM);
}
